// graphconv_16166256902194
// MI455X (gfx1250) — matrix operand order repaired and checked by reading, not compiled, not run on hardware
//
#include <hip/hip_runtime.h>
#include <hip/hip_bf16.h>

// ---------------------------------------------------------------------------
// MixProp graph conv for MI455X (gfx1250), bf16 WMMA with f32 accumulate.
//   x  : [B=8, C=32, N=512, T=168]  f32
//   adp: [512, 512]                 f32
//   W1/W2: [32, 96], b1/b2: [32]
// out = mixprop(x, norm(adp+I), W1, b1) + mixprop(x, norm(adpT+I), W2, b2)
//
// Memory-bound problem (~2 GB HBM traffic vs ~tens of us of bf16 WMMA math at
// 23.3 TB/s), so intermediates h1/h2 are stored as bf16: they are only ever
// consumed as WMMA B-operands, halving their read/write traffic. Adjacency
// (512 KB bf16) and the h buffers (42 MB) stay L2-resident (192 MB L2).
// ---------------------------------------------------------------------------

typedef __attribute__((ext_vector_type(16))) __bf16 v16bf;
typedef __attribute__((ext_vector_type(8)))  __bf16 v8bf;
typedef __attribute__((ext_vector_type(8)))  float  v8f;

#define GB   8
#define GC   32
#define GN   512
#define GT   168
#define GNT  (GN * GT)          // 86016 (divisible by 128)
#define GCIN 96
#define ALPHA_F 0.05f

__device__ __forceinline__ __bf16 f2bf(float f) { return (__bf16)f; }
__device__ __forceinline__ __bf16 tobf(float f)  { return (__bf16)f; }
__device__ __forceinline__ __bf16 tobf(__bf16 f) { return f; }

// ---------------------------------------------------------------------------
// Row sums of adp and adpT (plus the +1 from the added identity).
// ---------------------------------------------------------------------------
__global__ void rowcol_sum_kernel(const float* __restrict__ adp,
                                  float* __restrict__ rowsum,
                                  float* __restrict__ colsum) {
  __shared__ float sr[256];
  __shared__ float sc[256];
  const int v = blockIdx.x;
  const int t = threadIdx.x;
  float r = 0.f, c = 0.f;
  for (int w = t; w < GN; w += 256) {
    r += adp[v * GN + w];
    c += adp[w * GN + v];
  }
  sr[t] = r; sc[t] = c;
  __syncthreads();
  for (int s = 128; s > 0; s >>= 1) {
    if (t < s) { sr[t] += sr[t + s]; sc[t] += sc[t + s]; }
    __syncthreads();
  }
  if (t == 0) {
    rowsum[v] = sr[0] + 1.0f;   // + identity diagonal
    colsum[v] = sc[0] + 1.0f;
  }
}

// ---------------------------------------------------------------------------
// A1[v][w] = (adp[v][w] + d(v,w)) / rowsum[v]   (bf16)
// A2[v][w] = (adp[w][v] + d(v,w)) / colsum[v]   (bf16)
// ---------------------------------------------------------------------------
__global__ void build_adj_kernel(const float* __restrict__ adp,
                                 const float* __restrict__ rowsum,
                                 const float* __restrict__ colsum,
                                 __bf16* __restrict__ A1,
                                 __bf16* __restrict__ A2) {
  const int v = blockIdx.x;
  const float rinv = 1.0f / rowsum[v];
  const float cinv = 1.0f / colsum[v];
  for (int w = threadIdx.x; w < GN; w += blockDim.x) {
    const float d = (v == w) ? 1.0f : 0.0f;
    A1[v * GN + w] = f2bf((adp[v * GN + w] + d) * rinv);
    A2[v * GN + w] = f2bf((adp[w * GN + v] + d) * cinv);
  }
}

// ---------------------------------------------------------------------------
// Hout[s, n, t] = bf16( alpha*X[s,n,t] + (1-alpha)*sum_w A[n,w]*Hin[s,w,t] )
// HT = float (when Hin is x) or __bf16 (when Hin is a previous h buffer).
// One wave computes a 16(N) x 64(T) tile; A fragment reused across 4 WMMAs.
// T-edge handling: B-fragment column N=l only feeds accumulator column l,
// so out-of-range t columns just CLAMP the load address (garbage columns are
// never stored) -> branch-free inner loop, no EXEC gymnastics.
// grid = (ceil(T/64)=3, N/(16*8)=4, B*C=256), block = 256 (8 waves).
// ---------------------------------------------------------------------------
template <typename HT>
__global__ __launch_bounds__(256, 2)
void nconv_kernel(const __bf16* __restrict__ A,
                  const HT* __restrict__ Hin,
                  const float* __restrict__ X,
                  __bf16* __restrict__ Hout) {
  const int lane = threadIdx.x & 31;
  const int wave = threadIdx.x >> 5;
  const int l  = lane & 15;
  const int hi = lane >> 4;

  const int n0 = (blockIdx.y * 8 + wave) * 16;
  const int t0 = blockIdx.x * 64;
  const size_t slice = (size_t)blockIdx.z * GNT;

  // Clamped (always in-bounds) per-subtile column indices, loop-invariant.
  int tl[4];
#pragma unroll
  for (int sub = 0; sub < 4; ++sub) {
    const int t = t0 + sub * 16 + l;
    tl[sub] = (t < GT) ? t : (GT - 1);
  }

  v8f acc0 = {}, acc1 = {}, acc2 = {}, acc3 = {};

  // A fragment (contiguous k order, same as B): lanes 0-15 hold M=l, K = k0+{0..15};
  //                         lanes 16-31 hold M=l, K = k0+{16..31}.
  const __bf16* arow = A + (size_t)(n0 + l) * GN + hi * 16;
  const HT*     hbase = Hin + slice + (size_t)(hi * 16) * GT;

  for (int k0 = 0; k0 < GN; k0 += 32) {
    const v8bf alo = *(const v8bf*)(arow + k0);        // K = base .. base+7
    const v8bf ahi = *(const v8bf*)(arow + k0 + 8);    // K = base+8 .. base+15
    const v16bf af = __builtin_shufflevector(
        alo, ahi, 0, 1, 2, 3, 4, 5, 6, 7, 8, 9, 10, 11, 12, 13, 14, 15);

    // Branch-free prefetch of the next K tile (wraps to a valid address).
    const int kp = (k0 + 32) & (GN - 1);
    __builtin_prefetch(arow + kp, 0, 1);
    __builtin_prefetch(hbase + (size_t)kp * GT + tl[0], 0, 1);

    // B fragment (contiguous k order, same as A): lane holds column N=l, K = k0 + hi*16 + i.
    const HT* hk = hbase + (size_t)k0 * GT;
#pragma unroll
    for (int sub = 0; sub < 4; ++sub) {
      const HT* hp = hk + tl[sub];
      v16bf bf_;
#pragma unroll
      for (int i = 0; i < 16; ++i) bf_[i] = tobf(hp[(size_t)i * GT]);

      if (sub == 0)
        acc0 = __builtin_amdgcn_wmma_f32_16x16x32_bf16(false, af, false, bf_,
                                                       (short)0, acc0, false, false);
      else if (sub == 1)
        acc1 = __builtin_amdgcn_wmma_f32_16x16x32_bf16(false, af, false, bf_,
                                                       (short)0, acc1, false, false);
      else if (sub == 2)
        acc2 = __builtin_amdgcn_wmma_f32_16x16x32_bf16(false, af, false, bf_,
                                                       (short)0, acc2, false, false);
      else
        acc3 = __builtin_amdgcn_wmma_f32_16x16x32_bf16(false, af, false, bf_,
                                                       (short)0, acc3, false, false);
    }
  }

  // C/D layout: VGPR r -> M = hi*8 + r, column N = l.
  auto store_tile = [&](const v8f acc, int sub) {
    const int t = t0 + sub * 16 + l;
    if (t >= GT) return;
#pragma unroll
    for (int r = 0; r < 8; ++r) {
      const int row = n0 + hi * 8 + r;
      const size_t idx = slice + (size_t)row * GT + t;
      Hout[idx] = f2bf(ALPHA_F * X[idx] + (1.0f - ALPHA_F) * acc[r]);
    }
  };
  store_tile(acc0, 0);
  store_tile(acc1, 1);
  store_tile(acc2, 2);
  store_tile(acc3, 3);
}

// ---------------------------------------------------------------------------
// Shared chanmix helpers: one 16(o) x 16(m) WMMA per o-tile, K = C = 32.
// B fragment: H[c, m] tile -> lane = column m0+l, K = c = hi*16 + i.
// A fragment: weights W[o, cbase + k] (16-bit A layout as in nconv).
// ---------------------------------------------------------------------------

// out[b, o, m] = sum_c (W1[o,c] + W2[o,c]) * x[b, c, m] + b1[o] + b2[o]
// (the x term and biases are shared by both mixprop passes)
__global__ __launch_bounds__(256, 2)
void chanmix_init_kernel(const float* __restrict__ Xf,
                         const float* __restrict__ W1,
                         const float* __restrict__ W2,
                         const float* __restrict__ bias1,
                         const float* __restrict__ bias2,
                         float* __restrict__ Out) {
  const int lane = threadIdx.x & 31;
  const int wave = threadIdx.x >> 5;
  const int l  = lane & 15;
  const int hi = lane >> 4;

  const int b  = blockIdx.y;
  const int m0 = (blockIdx.x * 8 + wave) * 16;
  const size_t hbase = (size_t)b * GC * GNT;
  const size_t obase = (size_t)b * 32 * GNT;

  v16bf bfrag;
  {
    const float* hp = Xf + hbase + (size_t)(hi * 16) * GNT + (m0 + l);
#pragma unroll
    for (int i = 0; i < 16; ++i) bfrag[i] = f2bf(hp[(size_t)i * GNT]);
  }

#pragma unroll
  for (int ot = 0; ot < 2; ++ot) {
    const int o = ot * 16 + l;
    v16bf afrag;
#pragma unroll
    for (int i = 0; i < 16; ++i) {
      const int k = hi * 16 + i;
      const int wi = o * GCIN + k;                 // cbase = 0
      afrag[i] = f2bf(W1[wi] + W2[wi]);
    }
    v8f acc = {};
    acc = __builtin_amdgcn_wmma_f32_16x16x32_bf16(
        false, afrag, false, bfrag, (short)0, acc, false, false);

#pragma unroll
    for (int r = 0; r < 8; ++r) {
      const int oo = ot * 16 + hi * 8 + r;
      const size_t idx = obase + (size_t)oo * GNT + (m0 + l);
      Out[idx] = acc[r] + bias1[oo] + bias2[oo];
    }
  }
}

// out[b, o, m] += sum_c W[o, cbase + c] * H[b, c, m]   (H in bf16)
__global__ __launch_bounds__(256, 2)
void chanmix_acc_kernel(const __bf16* __restrict__ H,
                        const float* __restrict__ W,
                        int cbase,
                        float* __restrict__ Out) {
  const int lane = threadIdx.x & 31;
  const int wave = threadIdx.x >> 5;
  const int l  = lane & 15;
  const int hi = lane >> 4;

  const int b  = blockIdx.y;
  const int m0 = (blockIdx.x * 8 + wave) * 16;
  const size_t hbase = (size_t)b * GC * GNT;
  const size_t obase = (size_t)b * 32 * GNT;

  v16bf bfrag;
  {
    const __bf16* hp = H + hbase + (size_t)(hi * 16) * GNT + (m0 + l);
#pragma unroll
    for (int i = 0; i < 16; ++i) bfrag[i] = hp[(size_t)i * GNT];
  }

#pragma unroll
  for (int ot = 0; ot < 2; ++ot) {
    const int o = ot * 16 + l;
    v16bf afrag;
#pragma unroll
    for (int i = 0; i < 16; ++i) {
      const int k = hi * 16 + i;
      afrag[i] = f2bf(W[o * GCIN + cbase + k]);
    }
    v8f acc = {};
    acc = __builtin_amdgcn_wmma_f32_16x16x32_bf16(
        false, afrag, false, bfrag, (short)0, acc, false, false);

#pragma unroll
    for (int r = 0; r < 8; ++r) {
      const int oo = ot * 16 + hi * 8 + r;
      const size_t idx = obase + (size_t)oo * GNT + (m0 + l);
      Out[idx] += acc[r];   // exclusive element ownership per launch -> no atomics
    }
  }
}

// ---------------------------------------------------------------------------
extern "C" void kernel_launch(void* const* d_in, const int* in_sizes, int n_in,
                              void* d_out, int out_size, void* d_ws, size_t ws_size,
                              hipStream_t stream) {
  (void)in_sizes; (void)n_in; (void)out_size; (void)ws_size;

  const float* x   = (const float*)d_in[0];
  const float* adp = (const float*)d_in[1];
  const float* W1  = (const float*)d_in[2];
  const float* b1  = (const float*)d_in[3];
  const float* W2  = (const float*)d_in[4];
  const float* b2  = (const float*)d_in[5];
  float* out = (float*)d_out;

  // Workspace layout (~86 MB):
  //   [0,2KB)       rowsum        [2KB,4KB)   colsum
  //   [4KB, +512KB) A1 bf16       next 512KB  A2 bf16
  //   two bf16 ping-pong h buffers of B*C*N*T = 22,020,096 elements each.
  char* ws = (char*)d_ws;
  float*  rowsum = (float*)(ws);
  float*  colsum = (float*)(ws + 2048);
  __bf16* A1     = (__bf16*)(ws + 4096);
  __bf16* A2     = (__bf16*)(ws + 4096 + (size_t)GN * GN * 2);
  __bf16* buf0   = (__bf16*)(ws + 4096 + (size_t)GN * GN * 4);
  __bf16* buf1   = buf0 + (size_t)GB * GC * GNT;

  rowcol_sum_kernel<<<GN, 256, 0, stream>>>(adp, rowsum, colsum);
  build_adj_kernel<<<GN, 256, 0, stream>>>(adp, rowsum, colsum, A1, A2);

  const dim3 ngrid((GT + 63) / 64, GN / 128, GB * GC);  // (3, 4, 256)
  const dim3 cgrid(GNT / 128, GB);                      // (672, 8)

  // out = (W1[:,0:32] + W2[:,0:32]) @ x + (b1 + b2)   [shared x term of both passes]
  chanmix_init_kernel<<<cgrid, 256, 0, stream>>>(x, W1, W2, b1, b2, out);

  // ---- pass 1: forward adjacency ----
  nconv_kernel<float><<<ngrid, 256, 0, stream>>>(A1, x, x, buf0);       // h1
  chanmix_acc_kernel<<<cgrid, 256, 0, stream>>>(buf0, W1, 32, out);
  nconv_kernel<__bf16><<<ngrid, 256, 0, stream>>>(A1, buf0, x, buf1);   // h2
  chanmix_acc_kernel<<<cgrid, 256, 0, stream>>>(buf1, W1, 64, out);

  // ---- pass 2: transposed adjacency ----
  nconv_kernel<float><<<ngrid, 256, 0, stream>>>(A2, x, x, buf0);       // h1
  chanmix_acc_kernel<<<cgrid, 256, 0, stream>>>(buf0, W2, 32, out);
  nconv_kernel<__bf16><<<ngrid, 256, 0, stream>>>(A2, buf0, x, buf1);   // h2
  chanmix_acc_kernel<<<cgrid, 256, 0, stream>>>(buf1, W2, 64, out);
}
